// LlamaAttention_3453153706098
// MI455X (gfx1250) — matrix operand order repaired and checked by reading, not compiled, not run on hardware
//
#include <hip/hip_runtime.h>
#include <hip/hip_bf16.h>

// ---------------------------------------------------------------------------
// LlamaAttention for MI455X (gfx1250): bf16 WMMA pipeline + TDM-staged attention.
//   B=2, T=2048, C=2048, H=32, KV=8 (g=4), D=64.
// Matmuls via v_wmma_f32_16x16x32_bf16 (fp32 accumulate); K/V tiles staged to
// LDS with tensor_load_to_lds (TENSORcnt double-buffering).  The causal mask
// is applied only in the single diagonal block, peeled out of the hot loop.
// ---------------------------------------------------------------------------

typedef __attribute__((ext_vector_type(16))) __bf16 v16bf;
typedef __attribute__((ext_vector_type(8)))  __bf16 v8bf;
typedef __attribute__((ext_vector_type(8)))  float  v8f;
typedef __attribute__((ext_vector_type(4)))  unsigned int u32x4;
typedef __attribute__((ext_vector_type(8)))  unsigned int u32x8;

#define WMMA_BF16(A, Bm, Cc) \
  __builtin_amdgcn_wmma_f32_16x16x32_bf16(false, (A), false, (Bm), (short)0, (Cc), false, false)

#define CAT16(lo, hi) \
  __builtin_shufflevector((lo), (hi), 0, 1, 2, 3, 4, 5, 6, 7, 8, 9, 10, 11, 12, 13, 14, 15)

static constexpr int B_  = 2;
static constexpr int T_  = 2048;
static constexpr int C_  = 2048;
static constexpr int H_  = 32;
static constexpr int KV_ = 8;
static constexpr int D_  = 64;

__device__ __forceinline__ __bf16 f2bf(float f) {
  unsigned u = __builtin_bit_cast(unsigned, f);
  unsigned r = (u + 0x7FFFu + ((u >> 16) & 1u)) >> 16;
  unsigned short s = (unsigned short)r;
  return __builtin_bit_cast(__bf16, s);
}
__device__ __forceinline__ float bf2f(__bf16 b) {
  unsigned short s = __builtin_bit_cast(unsigned short, b);
  unsigned u = ((unsigned)s) << 16;
  return __builtin_bit_cast(float, u);
}

__device__ __forceinline__ void store_out(float* p, float v)  { *p = v; }
__device__ __forceinline__ void store_out(__bf16* p, float v) { *p = f2bf(v); }

// --------------------- TDM: 2-D bf16 tile -> LDS (async) -------------------
__device__ __forceinline__ void tdm_load_2d_bf16(unsigned lds_byte_off,
                                                 const void* gptr,
                                                 unsigned tile_d0, unsigned tile_d1,
                                                 unsigned tensor_d0, unsigned tensor_d1,
                                                 unsigned long long stride0_elems) {
  unsigned long long ga = (unsigned long long)gptr;
  u32x4 g0;
  g0[0] = 1u;                                                 // count=1 (valid)
  g0[1] = lds_byte_off;                                       // lds_addr
  g0[2] = (unsigned)(ga & 0xFFFFFFFFu);                       // global_addr[31:0]
  g0[3] = (unsigned)((ga >> 32) & 0x01FFFFFFu) | (2u << 30);  // [56:32] | type=2
  u32x8 g1;
  g1[0] = (1u << 16);                                         // data_size = 2 bytes
  g1[1] = (tensor_d0 & 0xFFFFu) << 16;                        // tensor_dim0[15:0]
  g1[2] = ((tensor_d0 >> 16) & 0xFFFFu) | ((tensor_d1 & 0xFFFFu) << 16);
  g1[3] = ((tensor_d1 >> 16) & 0xFFFFu) | ((tile_d0 & 0xFFFFu) << 16);
  g1[4] = (tile_d1 & 0xFFFFu);                                // tile_dim2 = 0
  g1[5] = (unsigned)(stride0_elems & 0xFFFFFFFFu);            // dim0 stride lo
  g1[6] = (unsigned)((stride0_elems >> 32) & 0xFFFFu);        // dim0 stride hi
  g1[7] = 0u;
  asm volatile("tensor_load_to_lds %0, %1" :: "s"(g0), "s"(g1) : "memory");
}

// ------------------------------ fp32 -> bf16 -------------------------------
__global__ void cvt_f32_bf16(const float* __restrict__ in, __bf16* __restrict__ out, int n) {
  int i = blockIdx.x * blockDim.x + threadIdx.x;
  if (i < n) out[i] = f2bf(in[i]);
}

// ---------------------- Generic WMMA GEMM: Y = X * W^T ---------------------
// X: [M x K] bf16 row-major, W: [N x K] bf16 row-major, Y: [M x N].
// One wave per block, 64x64 output tile: 16 WMMAs per 16 b128 loads per k-step.
template <typename OutT>
__global__ void __launch_bounds__(32)
gemm_bf16_wmma(const __bf16* __restrict__ X, const __bf16* __restrict__ W,
               OutT* __restrict__ Y, int K, int N) {
  const int lane = threadIdx.x;
  const int hf = lane >> 4;   // half-wave selector
  const int nn = lane & 15;
  const long m0 = (long)blockIdx.x * 64;
  const long n0 = (long)blockIdx.y * 64;

  v8f acc[4][4] = {};
  const __bf16* xr[4];
  const __bf16* wr[4];
#pragma unroll
  for (int i = 0; i < 4; ++i) {
    xr[i] = X + (m0 + i * 16 + nn) * (long)K;
    wr[i] = W + (n0 + i * 16 + nn) * (long)K;
  }

  for (int k0 = 0; k0 < K; k0 += 32) {
    // pull future K-slabs toward the WGP (global_prefetch_b8, speculative)
    __builtin_prefetch(xr[0] + k0 + 256, 0, 0);
    __builtin_prefetch(wr[0] + k0 + 256, 0, 0);
    // A tiles (16x32): lane row m = nn; halves hold K {0..7,16..23}/{8..15,24..31}
    v16bf a[4];
#pragma unroll
    for (int i = 0; i < 4; ++i) {
      v8bf lo = *(const v8bf*)(xr[i] + k0 + hf * 8);
      v8bf hi = *(const v8bf*)(xr[i] + k0 + 16 + hf * 8);
      a[i] = CAT16(lo, hi);
    }
    // B tiles (32x16): lane col n = nn; halves hold K {0..7,16..23}/{8..15,24..31}
    v16bf bm[4];
#pragma unroll
    for (int j = 0; j < 4; ++j)
      bm[j] = CAT16(*(const v8bf*)(wr[j] + k0 + hf * 8), *(const v8bf*)(wr[j] + k0 + 16 + hf * 8));
#pragma unroll
    for (int i = 0; i < 4; ++i)
#pragma unroll
      for (int j = 0; j < 4; ++j)
        acc[i][j] = WMMA_BF16(a[i], bm[j], acc[i][j]);
  }
  // C layout: VGPR r -> row tile_m + r + 8*hf, col = tile_n + nn.
#pragma unroll
  for (int i = 0; i < 4; ++i)
#pragma unroll
    for (int r = 0; r < 8; ++r) {
      long row = m0 + i * 16 + r + 8 * hf;
      OutT* yp = Y + row * (long)N + n0 + nn;
      store_out(yp +  0, acc[i][0][r]);
      store_out(yp + 16, acc[i][1][r]);
      store_out(yp + 32, acc[i][2][r]);
      store_out(yp + 48, acc[i][3][r]);
    }
}

// ------------------------------- RoPE + layout -----------------------------
__global__ void rope_q_kernel(const __bf16* __restrict__ qlin,
                              const float* __restrict__ cosb,
                              const float* __restrict__ sinb,
                              __bf16* __restrict__ qr) {
  int i = blockIdx.x * blockDim.x + threadIdx.x;     // B*T*H*32 threads
  if (i >= B_ * T_ * H_ * 32) return;
  int d = i & 31; int tmp = i >> 5;
  int h = tmp % H_; tmp /= H_;
  int t = tmp % T_; int b = tmp / T_;
  const __bf16* src = qlin + ((long)(b * T_ + t) * H_ + h) * 64;
  float x1 = bf2f(src[d]), x2 = bf2f(src[d + 32]);
  float c1 = cosb[t * 64 + d],      s1 = sinb[t * 64 + d];
  float c2 = cosb[t * 64 + d + 32], s2 = sinb[t * 64 + d + 32];
  float o1 = (x1 * c1 - x2 * s1) * 0.125f;   // 1/sqrt(64) folded into q
  float o2 = (x2 * c2 + x1 * s2) * 0.125f;
  __bf16* dst = qr + ((long)(b * H_ + h) * T_ + t) * 64;
  dst[d] = f2bf(o1); dst[d + 32] = f2bf(o2);
}

__global__ void rope_k_kernel(const __bf16* __restrict__ klin,
                              const float* __restrict__ cosb,
                              const float* __restrict__ sinb,
                              __bf16* __restrict__ kr) {
  int i = blockIdx.x * blockDim.x + threadIdx.x;     // B*T*KV*32 threads
  if (i >= B_ * T_ * KV_ * 32) return;
  int d = i & 31; int tmp = i >> 5;
  int kv = tmp % KV_; tmp /= KV_;
  int t = tmp % T_; int b = tmp / T_;
  const __bf16* src = klin + ((long)(b * T_ + t) * KV_ + kv) * 64;
  float x1 = bf2f(src[d]), x2 = bf2f(src[d + 32]);
  float c1 = cosb[t * 64 + d],      s1 = sinb[t * 64 + d];
  float c2 = cosb[t * 64 + d + 32], s2 = sinb[t * 64 + d + 32];
  __bf16* dst = kr + ((long)(b * KV_ + kv) * T_ + t) * 64;
  dst[d]      = f2bf(x1 * c1 - x2 * s1);
  dst[d + 32] = f2bf(x2 * c2 + x1 * s2);
}

__global__ void v_transpose_kernel(const __bf16* __restrict__ vlin,
                                   __bf16* __restrict__ vt) {
  int i = blockIdx.x * blockDim.x + threadIdx.x;     // B*T*KV*64 threads
  if (i >= B_ * T_ * KV_ * 64) return;
  int d = i & 63; int tmp = i >> 6;
  int kv = tmp % KV_; tmp /= KV_;
  int t = tmp % T_; int b = tmp / T_;
  vt[(((long)b * KV_ + kv) * 64 + d) * T_ + t] =
      vlin[((long)(b * T_ + t) * KV_ + kv) * 64 + d];
}

// --------------------------- Flash attention core --------------------------
// Block body expanded textually (macro) so LDS accesses stay lexically on
// `smem` (clean ds_load codegen) and the masked variant exists exactly once,
// outside the hot loop.  MASKED is a literal 0/1 -> mask code constant-folds.
#define ATTN_PROCESS_BLOCK(S0, BUF, MASKED)                                    \
  {                                                                            \
    const __bf16* lk = &smem[(BUF)][0];                                        \
    const __bf16* lv = &smem[(BUF)][2048];                                     \
    /* S^T tiles from LDS: rows = keys, cols = query rows */                   \
    v16bf ka0a, ka0b, ka1a, ka1b;                                              \
    {                                                                          \
      const __bf16* k0r = lk + nn * 64;                                        \
      const __bf16* k1r = lk + (16 + nn) * 64;                                 \
      v8bf x0 = *(const v8bf*)(k0r + hf * 8);                                  \
      v8bf x1 = *(const v8bf*)(k0r + 16 + hf * 8);                             \
      v8bf x2 = *(const v8bf*)(k0r + 32 + hf * 8);                             \
      v8bf x3 = *(const v8bf*)(k0r + 48 + hf * 8);                             \
      v8bf y0 = *(const v8bf*)(k1r + hf * 8);                                  \
      v8bf y1 = *(const v8bf*)(k1r + 16 + hf * 8);                             \
      v8bf y2 = *(const v8bf*)(k1r + 32 + hf * 8);                             \
      v8bf y3 = *(const v8bf*)(k1r + 48 + hf * 8);                             \
      ka0a = CAT16(x0, x1); ka0b = CAT16(x2, x3);                              \
      ka1a = CAT16(y0, y1); ka1b = CAT16(y2, y3);                              \
    }                                                                          \
    v8f c0 = {}, c1 = {};                                                      \
    c0 = WMMA_BF16(ka0a, qb0, c0);                                             \
    c0 = WMMA_BF16(ka0b, qb1, c0);                                             \
    c1 = WMMA_BF16(ka1a, qb0, c1);                                             \
    c1 = WMMA_BF16(ka1b, qb1, c1);                                             \
    /* (masked) online softmax over keys, per query column nn */               \
    float p0[8], p1[8];                                                        \
    float mx = -3.0e38f;                                                       \
    _Pragma("unroll")                                                          \
    for (int r = 0; r < 8; ++r) {                                              \
      float v0 = c0[r], v1 = c1[r];                                            \
      if (MASKED) {                                                            \
        int key0 = (S0) + r + 8 * hf;                                          \
        int tcol = t0 + nn;                                                    \
        if (key0 > tcol)      v0 = -3.0e38f;                                   \
        if (key0 + 16 > tcol) v1 = -3.0e38f;                                   \
      }                                                                        \
      p0[r] = v0; p1[r] = v1;                                                  \
      mx = fmaxf(mx, fmaxf(v0, v1));                                           \
    }                                                                          \
    mx = fmaxf(mx, __shfl_xor(mx, 16, 32));                                    \
    float mnew  = fmaxf(mrun, mx);                                             \
    float alpha = __expf(mrun - mnew);                                         \
    float ps = 0.0f;                                                           \
    _Pragma("unroll")                                                          \
    for (int r = 0; r < 8; ++r) {                                              \
      p0[r] = __expf(p0[r] - mnew);                                            \
      p1[r] = __expf(p1[r] - mnew);                                            \
      ps += p0[r] + p1[r];                                                     \
    }                                                                          \
    ps += __shfl_xor(ps, 16, 32);                                              \
    lrun = lrun * alpha + ps;                                                  \
    mrun = mnew;                                                               \
    o0 *= alpha; o1 *= alpha; o2 *= alpha; o3 *= alpha;                        \
    /* repack P^T (C layout) into WMMA B layout via shfl_xor(16) */            \
    v16bf pb;                                                                  \
    _Pragma("unroll")                                                          \
    for (int r = 0; r < 8; ++r) {                                              \
      float t0o = __shfl_xor(p0[r], 16, 32);                                   \
      float t1o = __shfl_xor(p1[r], 16, 32);                                   \
      float elo = (hf == 0) ? p0[r] : t1o;                                     \
      float ehi = (hf == 0) ? t0o : p1[r];                                     \
      pb[r]     = f2bf(elo);                                                   \
      pb[8 + r] = f2bf(ehi);                                                   \
    }                                                                          \
    /* out^T += V^T * P^T  (4 dim-tiles of 16), V^T from LDS */                \
    {                                                                          \
      const __bf16* vr0 = lv + ( 0 + nn) * 32;                                 \
      const __bf16* vr1 = lv + (16 + nn) * 32;                                 \
      const __bf16* vr2 = lv + (32 + nn) * 32;                                 \
      const __bf16* vr3 = lv + (48 + nn) * 32;                                 \
      v16bf va0 = *(const v16bf*)(vr0 + hf * 16);                              \
      v16bf va1 = *(const v16bf*)(vr1 + hf * 16);                              \
      v16bf va2 = *(const v16bf*)(vr2 + hf * 16);                              \
      v16bf va3 = *(const v16bf*)(vr3 + hf * 16);                              \
      o0 = WMMA_BF16(va0, pb, o0);                                             \
      o1 = WMMA_BF16(va1, pb, o1);                                             \
      o2 = WMMA_BF16(va2, pb, o2);                                             \
      o3 = WMMA_BF16(va3, pb, o3);                                             \
    }                                                                          \
  }

// One wave per (b, h, 16-query tile).  S^T = K*Q^T, out^T = V^T * P^T.
// K (32x64) and V^T (64x32) tiles DMA'd into LDS with tensor_load_to_lds,
// double-buffered on TENSORcnt.  Hot loop = unmasked blocks only; the single
// diagonal block is peeled after the loop.
__global__ void __launch_bounds__(32)
attn_fa_kernel(const __bf16* __restrict__ qr, const __bf16* __restrict__ kr,
               const __bf16* __restrict__ vt, __bf16* __restrict__ ao) {
  const int lane = threadIdx.x;
  const int hf = lane >> 4;
  const int nn = lane & 15;                 // query column within tile
  const int t0 = blockIdx.x * 16;
  const int h  = blockIdx.y;
  const int b  = blockIdx.z;
  const int kv = h / (H_ / KV_);

  // smem: per buffer 8KB = K tile [32][64] (4KB) + V^T tile [64][32] (4KB).
  // Single __shared__ allocation => LDS byte offset 0 for the D# descriptors.
  __shared__ __bf16 smem[2][4096];

  const __bf16* kbase = kr + ((long)b * KV_ + kv) * (long)T_ * 64;
  const __bf16* vbase = vt + ((long)b * KV_ + kv) * 64L * T_;

  // Q as B-operand (Q^T: K dim = head dim, col n = query row). Table order.
  const __bf16* qrow = qr + (((long)b * H_ + h) * T_ + (t0 + nn)) * 64;
  v16bf qb0 = CAT16(*(const v8bf*)(qrow + hf * 8), *(const v8bf*)(qrow + 16 + hf * 8));        // dims 0..31 slice
  v16bf qb1 = CAT16(*(const v8bf*)(qrow + 32 + hf * 8), *(const v8bf*)(qrow + 48 + hf * 8));   // dims 32..63 slice

  auto issue_block = [&](int buf, int s0) {
    unsigned base = (unsigned)buf * 8192u;
    // K tile: 32 rows of 64 bf16, row-major, stride 64 elems.
    tdm_load_2d_bf16(base, kbase + (long)s0 * 64, 64, 32, 64, 32, 64ull);
    // V^T tile: 64 rows of 32 bf16, row stride T elems.
    tdm_load_2d_bf16(base + 4096u, vbase + s0, 32, 64, 32, 64, (unsigned long long)T_);
  };

  issue_block(0, 0);

  float mrun = -3.0e38f, lrun = 0.0f;
  v8f o0 = {}, o1 = {}, o2 = {}, o3 = {};

  const int nb = (t0 + 16 + 31) / 32;       // total 32-key blocks (>= 1)
  int buf = 0;
  // ---- full (unmasked) blocks: no branch, no mask VALU in the hot loop ----
  for (int ib = 0; ib < nb - 1; ++ib, buf ^= 1) {
    const int s0 = ib * 32;
    issue_block(buf ^ 1, s0 + 32);
    __builtin_amdgcn_s_wait_tensorcnt(2);   // block ib done; ib+1 in flight
    ATTN_PROCESS_BLOCK(s0, buf, 0)
  }
  // ---- final diagonal block (causally masked), peeled ----
  {
    const int s0 = (nb - 1) * 32;
    __builtin_amdgcn_s_wait_tensorcnt(0);
    ATTN_PROCESS_BLOCK(s0, buf, 1)
  }

  // ---- normalize and store 16x64 tile to ao [b*T + t][h*64 + d] ----
  float inv = 1.0f / lrun;
  __bf16* aop = ao + ((long)b * T_ + t0 + nn) * (H_ * 64) + h * 64;
#pragma unroll
  for (int r = 0; r < 8; ++r) {
    int d = r + 8 * hf;
    aop[ 0 + d] = f2bf(o0[r] * inv);
    aop[16 + d] = f2bf(o1[r] * inv);
    aop[32 + d] = f2bf(o2[r] * inv);
    aop[48 + d] = f2bf(o3[r] * inv);
  }
}

// ---------------------------------------------------------------------------
extern "C" void kernel_launch(void* const* d_in, const int* in_sizes, int n_in,
                              void* d_out, int out_size, void* d_ws, size_t ws_size,
                              hipStream_t stream) {
  (void)in_sizes; (void)n_in; (void)out_size; (void)ws_size;
  const float* x    = (const float*)d_in[0];   // [B,T,C]
  const float* Wq   = (const float*)d_in[1];   // [H*D, C]
  const float* Wk   = (const float*)d_in[2];   // [KV*D, C]
  const float* Wv   = (const float*)d_in[3];   // [KV*D, C]
  const float* Wo   = (const float*)d_in[4];   // [C, H*D]
  const float* cosb = (const float*)d_in[5];   // [T, D]
  const float* sinb = (const float*)d_in[6];   // [T, D]
  float* out = (float*)d_out;                  // [B,T,C]

  const long M = (long)B_ * T_;                // 4096
  char* ws = (char*)d_ws;
  size_t off = 0;
  auto alloc = [&](size_t bytes) -> __bf16* {
    __bf16* p = (__bf16*)(ws + off);
    off += (bytes + 255) & ~(size_t)255;
    return p;
  };
  __bf16* xb   = alloc((size_t)M * C_ * 2);
  __bf16* wqb  = alloc((size_t)(H_ * D_) * C_ * 2);
  __bf16* wkb  = alloc((size_t)(KV_ * D_) * C_ * 2);
  __bf16* wvb  = alloc((size_t)(KV_ * D_) * C_ * 2);
  __bf16* wob  = alloc((size_t)C_ * (H_ * D_) * 2);
  __bf16* qlin = alloc((size_t)M * (H_ * D_) * 2);
  __bf16* klin = alloc((size_t)M * (KV_ * D_) * 2);
  __bf16* vlin = alloc((size_t)M * (KV_ * D_) * 2);
  __bf16* q_r  = alloc((size_t)M * (H_ * D_) * 2);
  __bf16* k_r  = alloc((size_t)M * (KV_ * D_) * 2);
  __bf16* v_t  = alloc((size_t)M * (KV_ * D_) * 2);
  __bf16* ao   = alloc((size_t)M * (H_ * D_) * 2);

  // 1) fp32 -> bf16 conversions
  {
    int n;
    n = (int)(M * C_);
    cvt_f32_bf16<<<n / 256, 256, 0, stream>>>(x, xb, n);
    n = H_ * D_ * C_;
    cvt_f32_bf16<<<n / 256, 256, 0, stream>>>(Wq, wqb, n);
    n = KV_ * D_ * C_;
    cvt_f32_bf16<<<n / 256, 256, 0, stream>>>(Wk, wkb, n);
    cvt_f32_bf16<<<n / 256, 256, 0, stream>>>(Wv, wvb, n);
    n = C_ * H_ * D_;
    cvt_f32_bf16<<<n / 256, 256, 0, stream>>>(Wo, wob, n);
  }

  // 2) QKV projections (Y = X * W^T), bf16 out
  gemm_bf16_wmma<__bf16><<<dim3((int)(M / 64), (H_ * D_) / 64), 32, 0, stream>>>(
      xb, wqb, qlin, C_, H_ * D_);
  gemm_bf16_wmma<__bf16><<<dim3((int)(M / 64), (KV_ * D_) / 64), 32, 0, stream>>>(
      xb, wkb, klin, C_, KV_ * D_);
  gemm_bf16_wmma<__bf16><<<dim3((int)(M / 64), (KV_ * D_) / 64), 32, 0, stream>>>(
      xb, wvb, vlin, C_, KV_ * D_);

  // 3) RoPE + head-major / transposed layouts
  rope_q_kernel<<<(B_ * T_ * H_ * 32) / 256, 256, 0, stream>>>(qlin, cosb, sinb, q_r);
  rope_k_kernel<<<(B_ * T_ * KV_ * 32) / 256, 256, 0, stream>>>(klin, cosb, sinb, k_r);
  v_transpose_kernel<<<(B_ * T_ * KV_ * 64) / 256, 256, 0, stream>>>(vlin, v_t);

  // 4) flash attention with TDM-staged K/V: one wave per (b, h, 16-query tile)
  attn_fa_kernel<<<dim3(T_ / 16, H_, B_), 32, 0, stream>>>(q_r, k_r, v_t, ao);

  // 5) output projection -> fp32 d_out
  gemm_bf16_wmma<float><<<dim3((int)(M / 64), C_ / 64), 32, 0, stream>>>(
      ao, wob, out, H_ * D_, C_);
}
